// MultiHeadAttention_22368189677671
// MI455X (gfx1250) — matrix operand order repaired and checked by reading, not compiled, not run on hardware
//
#include <hip/hip_runtime.h>
#include <hip/hip_bf16.h>
#include <math.h>

typedef __bf16 bf16_t;
typedef __attribute__((ext_vector_type(16))) __bf16 v16bf;
typedef __attribute__((ext_vector_type(8)))  __bf16 v8bf;
typedef __attribute__((ext_vector_type(8)))  float  v8f;

#define HIDDEN_ 1024
#define NHEAD_  16
#define HDIM_   64

// ---------- gfx1250 async-to-LDS staging (guarded; falls back to sync copy) ----------
#if __has_builtin(__builtin_amdgcn_global_load_async_to_lds_b128)
#define HAVE_ASYNC_CP 1
#endif
#if __has_builtin(__builtin_amdgcn_s_wait_asynccnt)
#define HAVE_ASYNC_WAIT 1
#endif

#if defined(HAVE_ASYNC_CP)
typedef int v4i_vs __attribute__((vector_size(16)));
typedef __attribute__((address_space(1))) v4i_vs as1_v4i;   // global
typedef __attribute__((address_space(3))) v4i_vs as3_v4i;   // LDS
#endif

static __device__ __forceinline__ void cp16_to_lds(bf16_t* lds_dst, const bf16_t* gsrc) {
#if defined(HAVE_ASYNC_CP)
  // flat->LDS offset is addr[31:0]; flat global address == global address (ISA 10.2)
  __builtin_amdgcn_global_load_async_to_lds_b128(
      (as1_v4i*)(uintptr_t)gsrc,
      (as3_v4i*)(uintptr_t)lds_dst, 0, 0);
#else
  *(v8bf*)lds_dst = *(const v8bf*)gsrc;
#endif
}

// Wait for this wave's outstanding async-LDS writes, then block barrier.
static __device__ __forceinline__ void staging_fence() {
#if defined(HAVE_ASYNC_CP) && defined(HAVE_ASYNC_WAIT)
  __builtin_amdgcn_s_wait_asynccnt(0);
#endif
  __syncthreads();
}

// ---------- helpers ----------

static __device__ __forceinline__ bf16_t f2bf(float f) {
  union { float f; unsigned u; } c; c.f = f;
  unsigned r = (c.u + 0x7FFFu + ((c.u >> 16) & 1u)) >> 16;   // RNE
  union { unsigned short s; bf16_t b; } o; o.s = (unsigned short)r;
  return o.b;
}

union FragU { v16bf v; v8bf h[2]; };

// A-operand (16x32, MxK), ISA 7.12.2: lane t -> row (t&15);
// K runs [half*8, +8) and [16+half*8, +8), half = t>>4.
static __device__ __forceinline__ v16bf load_a_frag(const bf16_t* __restrict__ base, int ld) {
  const int t = (int)(threadIdx.x & 31u);
  const bf16_t* p = base + (size_t)(t & 15) * ld + ((t >> 4) << 3);
  FragU f;
  f.h[0] = *(const v8bf*)(p);
  f.h[1] = *(const v8bf*)(p + 16);
  return f.v;
}

// B-operand (32x16, KxN) from K-contiguous [N][K] storage:
// lane t -> col N=(t&15); K runs [half*8, +8) and [16+half*8, +8).
static __device__ __forceinline__ v16bf load_b_frag(const bf16_t* __restrict__ base, int ld) {
  const int t = (int)(threadIdx.x & 31u);
  const bf16_t* p = base + (size_t)(t & 15) * ld + ((t >> 4) << 3);
  FragU f;
  f.h[0] = *(const v8bf*)(p);
  f.h[1] = *(const v8bf*)(p + 16);
  return f.v;
}

static __device__ __forceinline__ v8f wmma_bf16(v16bf a, v16bf b, v8f c) {
  return __builtin_amdgcn_wmma_f32_16x16x32_bf16(false, a, false, b, (short)0, c, false, false);
}

// Stage a 32-key x 64-d K chunk ([key][d], 4KB) -- 256 b128 transfers / 128 threads.
static __device__ __forceinline__ void stage_k(bf16_t* dst, const bf16_t* __restrict__ Kc,
                                               int kt, int h, int tid) {
  for (int i = tid; i < 256; i += 128) {
    int row = i >> 3, c = (i & 7) << 3;
    cp16_to_lds(&dst[row * 64 + c], Kc + (size_t)(kt + row) * HIDDEN_ + h * HDIM_ + c);
  }
}
// Stage a 64-d x 32-key V chunk ([d][key], 4KB).
static __device__ __forceinline__ void stage_v(bf16_t* dst, const bf16_t* __restrict__ Vt,
                                               int kt, int h, int tid, int S) {
  for (int i = tid; i < 256; i += 128) {
    int row = i >> 2, c = (i & 3) << 3;
    cp16_to_lds(&dst[row * 32 + c], Vt + (size_t)(h * HDIM_ + row) * S + kt + c);
  }
}

// ---------- kernels ----------

__global__ void cast_f32_to_bf16(const float* __restrict__ in, bf16_t* __restrict__ out, int n) {
  int i = blockIdx.x * blockDim.x + threadIdx.x;
  if (i < n) out[i] = f2bf(in[i]);
}

// C[M,N] = oscale * (A[M,K](bf16) @ W[N,K](bf16)^T + bias).
// Block = 4 waves; each wave: 32 rows x 64 cols (8 WMMA tiles, B-frags reused by 2 A-frags).
template <bool OUT_BF16, bool TRANSOUT>
__global__ void gemm_bias_kernel(const bf16_t* __restrict__ A,
                                 const bf16_t* __restrict__ W,
                                 const float*  __restrict__ bias,
                                 void* __restrict__ out,
                                 int M, int N, int K, float oscale) {
  const int wave = (int)(threadIdx.x >> 5);
  const int m0   = (blockIdx.x * 4 + wave) * 32;
  const int n0   = blockIdx.y * 64;
  const bf16_t* a0b = A + (size_t)m0 * K;
  const bf16_t* a1b = A + (size_t)(m0 + 16) * K;

  v8f acc[2][4] = {};
  for (int k0 = 0; k0 < K; k0 += 32) {
    if (k0 + 32 < K) {
      __builtin_prefetch((const void*)(a0b + k0 + 32), 0, 1);
      __builtin_prefetch((const void*)(a1b + k0 + 32), 0, 1);
    }
    v16bf a0 = load_a_frag(a0b + k0, K);
    v16bf a1 = load_a_frag(a1b + k0, K);
#pragma unroll
    for (int g = 0; g < 4; ++g) {
      v16bf b = load_b_frag(W + (size_t)(n0 + g * 16) * K + k0, K);
      acc[0][g] = wmma_bf16(a0, b, acc[0][g]);
      acc[1][g] = wmma_bf16(a1, b, acc[1][g]);
    }
  }

  const int t = (int)(threadIdx.x & 31u);
  const int n = t & 15, half = t >> 4;
#pragma unroll
  for (int mi = 0; mi < 2; ++mi) {
#pragma unroll
    for (int g = 0; g < 4; ++g) {
      const int col = n0 + g * 16 + n;
      const float bv = bias[col];
#pragma unroll
      for (int r = 0; r < 8; ++r) {
        const int row = m0 + mi * 16 + half * 8 + r;
        float v = (acc[mi][g][r] + bv) * oscale;
        if (OUT_BF16) {
          size_t idx = TRANSOUT ? (size_t)col * M + row : (size_t)row * N + col;
          ((bf16_t*)out)[idx] = f2bf(v);
        } else {
          ((float*)out)[(size_t)row * N + col] = v;
        }
      }
    }
  }
}

// Attention: block = 4 waves, 64 queries (16/wave), one head per block.y.
// Double-buffered async-to-LDS staging of K/V chunks overlaps global latency
// with WMMA compute. 1/sqrt(d) is pre-folded into Q. Double softmax composed
// analytically; per-lane online (m,l) with a single cross-lane merge.
__global__ void attn_kernel(const bf16_t* __restrict__ Q,
                            const bf16_t* __restrict__ Kc,   // [S][HIDDEN]
                            const bf16_t* __restrict__ Vt,   // [HIDDEN][S], d-major
                            bf16_t* __restrict__ Ctx,        // [S][HIDDEN]
                            int S) {
  __shared__ bf16_t kT[2][32 * 64];    // [buf][key][d]  8 KB
  __shared__ bf16_t vT[2][64 * 32];    // [buf][d][key]  8 KB
  __shared__ bf16_t pT[4][16 * 32];    // per-wave p2    4 KB
  const int h    = blockIdx.y;
  const int wave = (int)(threadIdx.x >> 5);
  const int tid  = (int)threadIdx.x;
  const int t = tid & 31;
  const int lanen = t & 15, half = t >> 4;
  const int qrow0 = (blockIdx.x * 4 + wave) * 16;

  const bf16_t* qbase = Q + (size_t)qrow0 * HIDDEN_ + h * HDIM_;
  const v16bf qa0 = load_a_frag(qbase, HIDDEN_);
  const v16bf qa1 = load_a_frag(qbase + 32, HIDDEN_);

  float m_[8], l_[8];
#pragma unroll
  for (int r = 0; r < 8; ++r) { m_[r] = -3.0e38f; l_[r] = 0.0f; }

  // ---- pass 1: per-lane online (max, expsum); double-buffered K staging ----
  stage_k(kT[0], Kc, 0, h, tid);
  int buf = 0;
  for (int kt = 0; kt < S; kt += 32, buf ^= 1) {
    staging_fence();                                  // cur buffer ready; prev reads done
    if (kt + 32 < S) stage_k(kT[buf ^ 1], Kc, kt + 32, h, tid);   // overlap with compute
    const bf16_t* kb = kT[buf];
    v8f s0 = {}, s1 = {};
    s0 = wmma_bf16(qa0, load_b_frag(kb, 64), s0);
    s0 = wmma_bf16(qa1, load_b_frag(kb + 32, 64), s0);
    s1 = wmma_bf16(qa0, load_b_frag(kb + 16 * 64, 64), s1);
    s1 = wmma_bf16(qa1, load_b_frag(kb + 16 * 64 + 32, 64), s1);
#pragma unroll
    for (int r = 0; r < 8; ++r) {
      float v0 = s0[r], v1 = s1[r];                  // scale already folded into Q
      float nm = fmaxf(m_[r], fmaxf(v0, v1));
      l_[r] = l_[r] * __expf(m_[r] - nm) + __expf(v0 - nm) + __expf(v1 - nm);
      m_[r] = nm;
    }
  }
  __syncthreads();                                    // all waves done with pass-1 buffers

  // pre-stage pass-2 chunk 0; its latency overlaps the (m,l) merge below
  stage_k(kT[0], Kc, 0, h, tid);
  stage_v(vT[0], Vt, 0, h, tid, S);

  // single cross-lane merge of (m,l) within each 16-lane half-group
#pragma unroll
  for (int r = 0; r < 8; ++r) {
#pragma unroll
    for (int off = 1; off < 16; off <<= 1) {
      float om = __shfl_xor(m_[r], off, 16);
      float ol = __shfl_xor(l_[r], off, 16);
      float nm = fmaxf(m_[r], om);
      l_[r] = l_[r] * __expf(m_[r] - nm) + ol * __expf(om - nm);
      m_[r] = nm;
    }
  }
  float invl[8], l2[8];
#pragma unroll
  for (int r = 0; r < 8; ++r) { invl[r] = 1.0f / l_[r]; l2[r] = 0.0f; }

  // ---- pass 2: p2 = exp(softmax1 - 1/l); accumulate l2 and O = p2 @ V ----
  v8f o0 = {}, o1 = {}, o2 = {}, o3 = {};
  buf = 0;
  for (int kt = 0; kt < S; kt += 32, buf ^= 1) {
    staging_fence();
    if (kt + 32 < S) {
      stage_k(kT[buf ^ 1], Kc, kt + 32, h, tid);
      stage_v(vT[buf ^ 1], Vt, kt + 32, h, tid, S);
    }
    const bf16_t* kb = kT[buf];
    const bf16_t* vb = vT[buf];

    v8f s0 = {}, s1 = {};
    s0 = wmma_bf16(qa0, load_b_frag(kb, 64), s0);
    s0 = wmma_bf16(qa1, load_b_frag(kb + 32, 64), s0);
    s1 = wmma_bf16(qa0, load_b_frag(kb + 16 * 64, 64), s1);
    s1 = wmma_bf16(qa1, load_b_frag(kb + 16 * 64 + 32, 64), s1);
#pragma unroll
    for (int r = 0; r < 8; ++r) {
      float p0 = __expf(__expf(s0[r] - m_[r]) * invl[r] - invl[r]);
      float p1 = __expf(__expf(s1[r] - m_[r]) * invl[r] - invl[r]);
      l2[r] += p0 + p1;
      pT[wave][(half * 8 + r) * 32 + lanen]      = f2bf(p0);
      pT[wave][(half * 8 + r) * 32 + 16 + lanen] = f2bf(p1);
    }
    v16bf a = load_a_frag(pT[wave], 32);              // per-wave tile; DS in-order per wave
    o0 = wmma_bf16(a, load_b_frag(vb + 0 * 16 * 32, 32), o0);
    o1 = wmma_bf16(a, load_b_frag(vb + 1 * 16 * 32, 32), o1);
    o2 = wmma_bf16(a, load_b_frag(vb + 2 * 16 * 32, 32), o2);
    o3 = wmma_bf16(a, load_b_frag(vb + 3 * 16 * 32, 32), o3);
  }

#pragma unroll
  for (int r = 0; r < 8; ++r) {
#pragma unroll
    for (int off = 1; off < 16; off <<= 1) l2[r] += __shfl_xor(l2[r], off, 16);
  }

  v8f oo[4] = {o0, o1, o2, o3};
#pragma unroll
  for (int g = 0; g < 4; ++g) {
#pragma unroll
    for (int r = 0; r < 8; ++r) {
      int row = qrow0 + half * 8 + r;
      int col = h * HDIM_ + g * 16 + lanen;
      Ctx[(size_t)row * HIDDEN_ + col] = f2bf(oo[g][r] / l2[r]);
    }
  }
}

// ---------- launcher ----------

extern "C" void kernel_launch(void* const* d_in, const int* in_sizes, int n_in,
                              void* d_out, int out_size, void* d_ws, size_t ws_size,
                              hipStream_t stream) {
  const float* x  = (const float*)d_in[0];
  const float* wq = (const float*)d_in[1];
  const float* bq = (const float*)d_in[2];
  const float* wk = (const float*)d_in[3];
  const float* bk = (const float*)d_in[4];
  const float* wv = (const float*)d_in[5];
  const float* bv = (const float*)d_in[6];
  const float* wo = (const float*)d_in[7];
  const float* bo = (const float*)d_in[8];

  const int S = in_sizes[0] / HIDDEN_;           // 4096
  const size_t SZX = (size_t)S * HIDDEN_;
  const size_t SZW = (size_t)HIDDEN_ * HIDDEN_;

  bf16_t* ws  = (bf16_t*)d_ws;
  bf16_t* xb  = ws;
  bf16_t* wqb = xb  + SZX;
  bf16_t* wkb = wqb + SZW;
  bf16_t* wvb = wkb + SZW;
  bf16_t* wob = wvb + SZW;
  bf16_t* Qb  = wob + SZW;
  bf16_t* Kb  = Qb  + SZX;
  bf16_t* Vt  = Kb  + SZX;     // [HIDDEN][S]
  bf16_t* Ctx = Vt  + SZX;

  cast_f32_to_bf16<<<(int)((SZX + 255) / 256), 256, 0, stream>>>(x,  xb,  (int)SZX);
  cast_f32_to_bf16<<<(int)((SZW + 255) / 256), 256, 0, stream>>>(wq, wqb, (int)SZW);
  cast_f32_to_bf16<<<(int)((SZW + 255) / 256), 256, 0, stream>>>(wk, wkb, (int)SZW);
  cast_f32_to_bf16<<<(int)((SZW + 255) / 256), 256, 0, stream>>>(wv, wvb, (int)SZW);
  cast_f32_to_bf16<<<(int)((SZW + 255) / 256), 256, 0, stream>>>(wo, wob, (int)SZW);

  dim3 gblk(128);
  dim3 ggrid(S / 128, HIDDEN_ / 64);
  // 1/sqrt(HEAD_DIM)=0.125 folded into Q projection output
  gemm_bias_kernel<true,  false><<<ggrid, gblk, 0, stream>>>(xb, wqb, bq, Qb, S, HIDDEN_, HIDDEN_, 0.125f);
  gemm_bias_kernel<true,  false><<<ggrid, gblk, 0, stream>>>(xb, wkb, bk, Kb, S, HIDDEN_, HIDDEN_, 1.0f);
  gemm_bias_kernel<true,  true ><<<ggrid, gblk, 0, stream>>>(xb, wvb, bv, Vt, S, HIDDEN_, HIDDEN_, 1.0f);

  attn_kernel<<<dim3(S / 64, NHEAD_), 128, 0, stream>>>(Qb, Kb, Vt, Ctx, S);

  gemm_bias_kernel<false, false><<<ggrid, gblk, 0, stream>>>(Ctx, wob, bo, d_out, S, HIDDEN_, HIDDEN_, 1.0f);
}
